// Attention_44959717654749
// MI455X (gfx1250) — matrix operand order repaired and checked by reading, not compiled, not run on hardware
//
#include <hip/hip_runtime.h>
#include <hip/hip_bf16.h>

typedef __bf16 bf16;
typedef __attribute__((ext_vector_type(16))) __bf16 v16bf;
typedef __attribute__((ext_vector_type(8)))  float  v8f;

// Exact pointee type from hipcc's diagnostic for the async-LDS builtin:
// "__attribute__((__vector_size__(4 * sizeof(int)))) int"
typedef int i32x4 __attribute__((vector_size(16)));
typedef __attribute__((address_space(1))) i32x4 gbl_i32x4;
typedef __attribute__((address_space(3))) i32x4 lds_i32x4;

#define D_MODEL   1024
#define NUM_HEADS 16
#define SEQ       2048
#define BATCH     2
#define HEAD_DIM  64

#if defined(__gfx1250__) && __has_builtin(__builtin_amdgcn_global_load_async_to_lds_b128)
#define HAVE_ASYNC_LDS 1
#endif

// ---------------------------------------------------------------------------
// Fragment gathers (contiguous k order, same for A and B, wave32)
// ---------------------------------------------------------------------------

// 16-bit A-matrix 16x32 (MxK). Caller selects the row (M = base + lane%16) and
// passes a pointer to that K-contiguous row. Lane<16 holds K 0..15,
// lane>=16 holds K 16..31; each VGPR packs a contiguous K pair.
static __device__ inline v16bf frag_a_row(const bf16* __restrict__ row, int lane) {
  const int koff = (lane >= 16) ? 16 : 0;
  v16bf a;
#pragma unroll
  for (int v = 0; v < 8; ++v) {
    const int kb = 2 * v + koff;
    a[2 * v]     = row[kb];
    a[2 * v + 1] = row[kb + 1];
  }
  return a;
}

// 16-bit B-matrix 32x16 (KxN). Lane holds column N = base + lane%16.
// Lanes 0..15 hold K=0..15, lanes 16..31 hold K=16..31, VGPR v packs K pair
// (2v,2v+1). Caller passes pointer to the column's K-run already offset by
// the lane-half's +16.
static __device__ inline v16bf frag_b_krun(const bf16* __restrict__ col) {
  v16bf b;
#pragma unroll
  for (int v = 0; v < 8; ++v) {
    b[2 * v]     = col[2 * v];
    b[2 * v + 1] = col[2 * v + 1];
  }
  return b;
}

// ---------------------------------------------------------------------------
// Generic C[M,N] = A[M,K] * B[N,K]^T  (torch Linear: y = x @ W.T)
//   mode 0: store f32 row-major [M,N] to outF
//   mode 1: store bf16 head-split [B,H,S,64] to outH  (n = h*64+dd, m = b*S+s)
// WG = 256 threads = 8 waves arranged 4(M) x 2(N); wave tile 32x64 (8 accum);
// WG tile 128x128; K-step 32 with f32->bf16 staging in LDS.
// Per k-step per wave: 8 WMMAs vs 6 fragment gathers.
// ---------------------------------------------------------------------------
__global__ __launch_bounds__(256)
void gemm_bt_kernel(const float* __restrict__ A, const float* __restrict__ Bw,
                    float* __restrict__ outF, unsigned short* __restrict__ outH,
                    int M, int N, int K, int mode) {
  __shared__ bf16 lA[128][34];  // [m][k], padded
  __shared__ bf16 lB[128][34];  // [n][k], padded (K-contiguous for B frags)

  const int t    = threadIdx.x;
  const int lane = t & 31;
  const int w    = t >> 5;
  const int wm   = w >> 1;      // 0..3 -> 32 rows each
  const int wn   = w & 1;       // 0..1 -> 64 cols each
  const int m0   = blockIdx.y * 128;
  const int n0   = blockIdx.x * 128;

  v8f acc[2][4];
#pragma unroll
  for (int i = 0; i < 2; ++i)
#pragma unroll
    for (int j = 0; j < 4; ++j) {
      v8f z = {};
      acc[i][j] = z;
    }

  // 128 rows x 32 cols / 256 threads -> 16 elements each (same map for A & B)
  const int srow = t >> 1, scol = (t & 1) * 16;

  for (int k0 = 0; k0 < K; k0 += 32) {
    __syncthreads();
    {
      const float* src = A + (size_t)(m0 + srow) * K + k0 + scol;
#pragma unroll
      for (int i = 0; i < 16; i += 4) {
        float4 f = *(const float4*)(src + i);
        lA[srow][scol + i + 0] = (bf16)f.x;
        lA[srow][scol + i + 1] = (bf16)f.y;
        lA[srow][scol + i + 2] = (bf16)f.z;
        lA[srow][scol + i + 3] = (bf16)f.w;
      }
      const float* srcb = Bw + (size_t)(n0 + srow) * K + k0 + scol;
#pragma unroll
      for (int i = 0; i < 16; i += 4) {
        float4 f = *(const float4*)(srcb + i);
        lB[srow][scol + i + 0] = (bf16)f.x;
        lB[srow][scol + i + 1] = (bf16)f.y;
        lB[srow][scol + i + 2] = (bf16)f.z;
        lB[srow][scol + i + 3] = (bf16)f.w;
      }
    }
    __syncthreads();

    v16bf af[2], bfr[4];
#pragma unroll
    for (int i = 0; i < 2; ++i)
      af[i] = frag_a_row(&lA[wm * 32 + i * 16 + (lane & 15)][0], lane);
#pragma unroll
    for (int j = 0; j < 4; ++j)
      bfr[j] = frag_b_krun(&lB[wn * 64 + j * 16 + (lane & 15)][(lane >= 16) ? 16 : 0]);

#pragma unroll
    for (int i = 0; i < 2; ++i)
#pragma unroll
      for (int j = 0; j < 4; ++j)
        acc[i][j] = __builtin_amdgcn_wmma_f32_16x16x32_bf16(
            false, af[i], false, bfr[j], (short)0, acc[i][j], false, false);
  }

  const int hi = (lane >= 16) ? 8 : 0;
  if (mode == 0) {
#pragma unroll
    for (int i = 0; i < 2; ++i)
#pragma unroll
      for (int j = 0; j < 4; ++j) {
        const int gn = n0 + wn * 64 + j * 16 + (lane & 15);
        const int gmb = m0 + wm * 32 + i * 16 + hi;
#pragma unroll
        for (int r = 0; r < 8; ++r)
          outF[(size_t)(gmb + r) * N + gn] = acc[i][j][r];
      }
  } else {
    bf16* ohp = (bf16*)outH;
#pragma unroll
    for (int i = 0; i < 2; ++i)
#pragma unroll
      for (int j = 0; j < 4; ++j) {
        const int gn = n0 + wn * 64 + j * 16 + (lane & 15);
        const int h = gn >> 6, dd = gn & (HEAD_DIM - 1);
#pragma unroll
        for (int r = 0; r < 8; ++r) {
          const int gm = m0 + wm * 32 + i * 16 + r + hi;
          const int b = gm >> 11, s = gm & (SEQ - 1);   // S = 2048
          const size_t idx =
              (((size_t)b * NUM_HEADS + h) * SEQ + s) * HEAD_DIM + dd;
          ohp[idx] = (bf16)acc[i][j][r];
        }
      }
  }
}

// ---------------------------------------------------------------------------
// Flash attention over head-split bf16 Q/K/V [B*H, S, 64].
// Grid: (S/128, B*H); 256 threads; wave w owns 16 query rows.
// Per j-block (32 keys): WG stages K[32][64] (async copy to LDS when the
// gfx1250 builtin exists; ASYNCcnt fenced before the barrier) and a
// transposed V^T[64][32]; each wave: S-tile = QK^T (2x16x16 chained WMMA),
// scale, online softmax (shfl_xor butterflies within 16-lane halves),
// P -> per-wave LDS -> A-frag, O(16x64) += P*V via 4 WMMAs.
// ---------------------------------------------------------------------------
__global__ __launch_bounds__(256)
void flash_attn_kernel(const unsigned short* __restrict__ Qr,
                       const unsigned short* __restrict__ Kr,
                       const unsigned short* __restrict__ Vr,
                       float* __restrict__ O) {
  __shared__ bf16 Klds[32][80];      // [j][d]; 160B row stride (16B aligned)
  __shared__ bf16 Vt[64][34];        // [d][j], j-contiguous (B-frag for PV)
  __shared__ bf16 Plds[8][16][34];   // per-wave P re-layout scratch

  const bf16* Q  = (const bf16*)Qr;
  const bf16* Kg = (const bf16*)Kr;
  const bf16* Vg = (const bf16*)Vr;

  const int t    = threadIdx.x;
  const int lane = t & 31;
  const int w    = t >> 5;
  const int bh   = blockIdx.y;
  const int q0   = blockIdx.x * 128 + w * 16;
  const size_t base = (size_t)bh * SEQ * HEAD_DIM;

  // Q A-fragments for d=0..31 and d=32..63, resident all kernel.
  v16bf qf[2];
  {
    const bf16* qrow = Q + base + (size_t)(q0 + (lane & 15)) * HEAD_DIM;
    const int koff = (lane >= 16) ? 16 : 0;
#pragma unroll
    for (int s = 0; s < 2; ++s) {
#pragma unroll
      for (int v = 0; v < 8; ++v) {
        const int kb = 2 * v + koff + 32 * s;
        qf[s][2 * v]     = qrow[kb];
        qf[s][2 * v + 1] = qrow[kb + 1];
      }
    }
  }

  float mrow[8], lrow[8];
#pragma unroll
  for (int r = 0; r < 8; ++r) { mrow[r] = -1e30f; lrow[r] = 0.0f; }
  v8f acco[4];
#pragma unroll
  for (int c = 0; c < 4; ++c) { v8f z = {}; acco[c] = z; }

  const float scale = 0.125f;  // 1/sqrt(64)
  const int jrow = t >> 3, dcol8 = (t & 7) * 8;

  for (int jb = 0; jb < SEQ; jb += 32) {
    __syncthreads();
    {
      const bf16* ksrc = Kg + base + (size_t)(jb + jrow) * HEAD_DIM + dcol8;
#ifdef HAVE_ASYNC_LDS
      // 16B per lane, global -> LDS without a VGPR round trip (ASYNCcnt).
      __builtin_amdgcn_global_load_async_to_lds_b128(
          (gbl_i32x4*)(void*)ksrc,
          (lds_i32x4*)&Klds[jrow][dcol8],
          0, 0);
#else
#pragma unroll
      for (int i = 0; i < 8; ++i) Klds[jrow][dcol8 + i] = ksrc[i];
#endif
      const bf16* vsrc = Vg + base + (size_t)(jb + jrow) * HEAD_DIM + dcol8;
#pragma unroll
      for (int i = 0; i < 8; ++i) Vt[dcol8 + i][jrow] = vsrc[i];
    }
#ifdef HAVE_ASYNC_LDS
    asm volatile("s_wait_asynccnt 0x0" ::: "memory");
#endif
    __syncthreads();

    // ---- scores: two 16x16 j-half tiles, K-dim = d (2 chained WMMAs) ----
    v8f sc[2];
#pragma unroll
    for (int tj = 0; tj < 2; ++tj) {
      v8f c = {};
#pragma unroll
      for (int s = 0; s < 2; ++s) {
        const bf16* kp =
            &Klds[tj * 16 + (lane & 15)][32 * s + ((lane >= 16) ? 16 : 0)];
        v16bf bk = frag_b_krun(kp);
        c = __builtin_amdgcn_wmma_f32_16x16x32_bf16(
            false, qf[s], false, bk, (short)0, c, false, false);
      }
      sc[tj] = c;
    }

    // ---- online softmax over the 32 new columns ----
#pragma unroll
    for (int r = 0; r < 8; ++r) {
      const float s0 = sc[0][r] * scale;
      const float s1 = sc[1][r] * scale;
      float vmax = fmaxf(s0, s1);
#pragma unroll
      for (int off = 8; off >= 1; off >>= 1)
        vmax = fmaxf(vmax, __shfl_xor(vmax, off, 32));
      const float nm = fmaxf(mrow[r], vmax);
      const float p0 = __expf(s0 - nm);
      const float p1 = __expf(s1 - nm);
      float psum = p0 + p1;
#pragma unroll
      for (int off = 8; off >= 1; off >>= 1)
        psum += __shfl_xor(psum, off, 32);
      const float alpha = __expf(mrow[r] - nm);
      lrow[r] = lrow[r] * alpha + psum;
      mrow[r] = nm;
#pragma unroll
      for (int c = 0; c < 4; ++c) acco[c][r] *= alpha;
      // stash P (C-layout -> LDS, re-gathered below in A-layout)
      const int prow = r + ((lane >= 16) ? 8 : 0);
      Plds[w][prow][(lane & 15)]      = (bf16)p0;
      Plds[w][prow][16 + (lane & 15)] = (bf16)p1;
    }
    // Compiler fence: per-wave DS ops execute in order in HW; just keep the
    // compiler from hoisting the A-frag reads above the stores.
    asm volatile("" ::: "memory");

    // ---- O(16x64) += P(16x32) * V(32x64): A-frag of P, 4 B-frags of V ----
    v16bf pf = frag_a_row(&Plds[w][lane & 15][0], lane);
#pragma unroll
    for (int c = 0; c < 4; ++c) {
      const bf16* vp = &Vt[c * 16 + (lane & 15)][(lane >= 16) ? 16 : 0];
      v16bf bv = frag_b_krun(vp);
      acco[c] = __builtin_amdgcn_wmma_f32_16x16x32_bf16(
          false, pf, false, bv, (short)0, acco[c], false, false);
    }
  }

  // ---- finalize: divide by row sums, store f32 [B, S, D] ----
  const int b = bh / NUM_HEADS, h = bh % NUM_HEADS;
#pragma unroll
  for (int c = 0; c < 4; ++c) {
#pragma unroll
    for (int r = 0; r < 8; ++r) {
      const int srow = q0 + r + ((lane >= 16) ? 8 : 0);
      const int dd   = c * 16 + (lane & 15);
      O[((size_t)b * SEQ + srow) * D_MODEL + h * HEAD_DIM + dd] =
          acco[c][r] / lrow[r];
    }
  }
}

// ---------------------------------------------------------------------------
// Launch: QKV projections (head-split bf16) -> flash attention -> out proj.
// Workspace: Q,K,V bf16 (8 MB each) + O f32 (16 MB) = 40 MB.
// ---------------------------------------------------------------------------
extern "C" void kernel_launch(void* const* d_in, const int* in_sizes, int n_in,
                              void* d_out, int out_size, void* d_ws, size_t ws_size,
                              hipStream_t stream) {
  (void)in_sizes; (void)n_in; (void)out_size; (void)ws_size;
  const float* x  = (const float*)d_in[0];
  // d_in[1] = e, unused by the reference forward.
  const float* wq = (const float*)d_in[2];
  const float* wk = (const float*)d_in[3];
  const float* wv = (const float*)d_in[4];
  const float* wo = (const float*)d_in[5];

  const size_t qkv_bytes = (size_t)BATCH * NUM_HEADS * SEQ * HEAD_DIM * 2;  // 8 MB
  char* ws = (char*)d_ws;
  unsigned short* Qb = (unsigned short*)(ws);
  unsigned short* Kb = (unsigned short*)(ws + qkv_bytes);
  unsigned short* Vb = (unsigned short*)(ws + 2 * qkv_bytes);
  float*          Ob = (float*)(ws + 3 * qkv_bytes);

  const int M = BATCH * SEQ;   // 4096
  const int N = D_MODEL;       // 1024
  const int K = D_MODEL;       // 1024

  dim3 gg(N / 128, M / 128);   // (8, 32)
  dim3 bb(256);

  gemm_bt_kernel<<<gg, bb, 0, stream>>>(x, wq, nullptr, Qb, M, N, K, 1);
  gemm_bt_kernel<<<gg, bb, 0, stream>>>(x, wk, nullptr, Kb, M, N, K, 1);
  gemm_bt_kernel<<<gg, bb, 0, stream>>>(x, wv, nullptr, Vb, M, N, K, 1);

  dim3 ga(SEQ / 128, BATCH * NUM_HEADS);  // (16, 32)
  flash_attn_kernel<<<ga, bb, 0, stream>>>(Qb, Kb, Vb, Ob);

  gemm_bt_kernel<<<gg, bb, 0, stream>>>(Ob, wo, (float*)d_out, nullptr, M, N, K, 0);
}
